// GeneratorWrapper_56762287784260
// MI455X (gfx1250) — matrix operand order repaired and checked by reading, not compiled, not run on hardware
//
#include <hip/hip_runtime.h>
#include <math.h>
#include <stdint.h>

// ---------------- model constants (from reference) ----------------
#define NLAYER   16
#define DMODEL   1024
#define VOCAB    32000
#define MEMLEN   512
#define TPROMPT  128
#define NGEN     256
#define TOPK     40
#define OUTW     (NLAYER * DMODEL)   // 16384 concatenated layer outputs

typedef _Float16 half16 __attribute__((ext_vector_type(16)));
typedef _Float16 half8  __attribute__((ext_vector_type(8)));
typedef _Float16 half4  __attribute__((ext_vector_type(4)));
typedef float    floatx8 __attribute__((ext_vector_type(8)));

// ---------------- WMMA helpers ----------------
__device__ __forceinline__ floatx8 wmma_f16(half16 a, half16 b, floatx8 c) {
  // v_wmma_f32_16x16x32_f16  (8-arg form, codegen-verified on gfx1250)
  return __builtin_amdgcn_wmma_f32_16x16x32_f16(false, a, false, b, (short)0, c, false, false);
}

// One wave computes out[n] = sum_{d<1024} h[d] * W[d*ld + n] for n = 0..15.
// W must already be offset to the first of the 16 columns.
// A layout (16-bit, 16x32): lanes 0-15 = M rows with K 0..7 / 16..23,
// lanes 16-31 same M rows with K 8..15 / 24..31.  We broadcast h into every
// M row (rows 1..15 produce ignored D rows).  B layout: lane%16 = N column,
// 16 elements hold the same K rows as A.
__device__ __forceinline__ float gemv16(const _Float16* __restrict__ W, int ld,
                                        const _Float16* __restrict__ h) {
  const int lane = threadIdx.x & 31;
  const int hsel = lane >> 4;
  floatx8 c = {0.f, 0.f, 0.f, 0.f, 0.f, 0.f, 0.f, 0.f};
#pragma unroll 4
  for (int k0 = 0; k0 < DMODEL; k0 += 32) {
    half8 alo = *(const half8*)(h + k0 + hsel * 8);
    half8 ahi = *(const half8*)(h + k0 + 16 + hsel * 8);
    half16 a = __builtin_shufflevector(alo, ahi, 0, 1, 2, 3, 4, 5, 6, 7,
                                       8, 9, 10, 11, 12, 13, 14, 15);
    half16 b;
#pragma unroll
    for (int s = 0; s < 16; ++s)
      b[s] = W[(size_t)(k0 + (s < 8 ? hsel * 8 + s : 16 + hsel * 8 + (s - 8))) * (size_t)ld + (lane & 15)];
    c = wmma_f16(a, b, c);
  }
  return c[0];   // D VGPR0, lanes 0-15 = (M=0, N=lane)
}

// ---------------- pre-pass kernels ----------------
__global__ void k_f32_to_f16(const float* __restrict__ src,
                             _Float16* __restrict__ dst, int n) {
  int stride = gridDim.x * blockDim.x * 4;
  for (int j = (blockIdx.x * blockDim.x + threadIdx.x) * 4; j < n; j += stride) {
    float4 v = *(const float4*)(src + j);
    half4 h;
    h[0] = (_Float16)v.x; h[1] = (_Float16)v.y;
    h[2] = (_Float16)v.z; h[3] = (_Float16)v.w;
    *(half4*)(dst + j) = h;
  }
}

__global__ void k_init_state(const float* __restrict__ mems_init,
                             const int* __restrict__ token_counts,
                             float* __restrict__ ctx, int* __restrict__ counts) {
  int idx = blockIdx.x * blockDim.x + threadIdx.x;
  if (idx < VOCAB) counts[idx] = token_counts[idx];
  if (idx < OUTW) {
    int l = idx >> 10, d = idx & 1023;
    // mems_init[l, MEMLEN-1, 0, d]
    ctx[idx] = mems_init[(size_t)l * MEMLEN * DMODEL + (size_t)(MEMLEN - 1) * DMODEL + d];
  }
}

__global__ void k_build_H(const int* __restrict__ input_ids,
                          const float* __restrict__ E, _Float16* __restrict__ H) {
  int idx = blockIdx.x * blockDim.x + threadIdx.x;   // 128*1024 threads
  int t = idx >> 10, d = idx & 1023;
  int tok = input_ids[t];                            // (128, B=1)
  H[idx] = (_Float16)E[(size_t)tok * DMODEL + d];
}

// ---------------- prompt GEMM: Z[t, l*1024+e] = H[t,:] . Wl[l][:,e] --------
__global__ void k_prompt_gemm(const _Float16* __restrict__ H,
                              const _Float16* __restrict__ WlB,
                              float* __restrict__ Z) {
  int wave = blockIdx.x * (blockDim.x >> 5) + (threadIdx.x >> 5); // 8192 waves
  int lane = threadIdx.x & 31;
  int hsel = lane >> 4, ll = lane & 15;
  int mt = wave >> 10;          // 8 M tiles (128 tokens)
  int nt = wave & 1023;         // 1024 N tiles (16384 cols)
  int l = nt >> 6;
  int n0 = (nt & 63) << 4;
  const _Float16* Wb = WlB + ((size_t)l << 20) + n0;
  const _Float16* Hb = H + (size_t)((mt << 4) + ll) * DMODEL;
  floatx8 c = {0.f, 0.f, 0.f, 0.f, 0.f, 0.f, 0.f, 0.f};
#pragma unroll 4
  for (int k0 = 0; k0 < DMODEL; k0 += 32) {
    half8 alo = *(const half8*)(Hb + k0 + hsel * 8);
    half8 ahi = *(const half8*)(Hb + k0 + 16 + hsel * 8);
    half16 a = __builtin_shufflevector(alo, ahi, 0, 1, 2, 3, 4, 5, 6, 7,
                                       8, 9, 10, 11, 12, 13, 14, 15);
    half16 b;
#pragma unroll
    for (int s = 0; s < 16; ++s)
      b[s] = Wb[((size_t)(k0 + (s < 8 ? hsel * 8 + s : 16 + hsel * 8 + (s - 8))) << 10) + ll];
    c = wmma_f16(a, b, c);
  }
  int col = (l << 10) + n0 + ll;
  int rbase = (mt << 4) + (hsel << 3);   // D VGPR r: M = r (+8 for upper lanes)
#pragma unroll
  for (int r = 0; r < 8; ++r)
    Z[(size_t)(rbase + r) * OUTW + col] = c[r];
}

// sequential tanh recurrence over the 128 prompt steps (elementwise)
__global__ void k_prompt_scan(const float* __restrict__ Z, float* __restrict__ ctx,
                              _Float16* __restrict__ h15) {
  int idx = blockIdx.x * blockDim.x + threadIdx.x;   // 16384
  float c = ctx[idx];
  for (int t = 0; t < TPROMPT; ++t)
    c = tanhf(Z[(size_t)t * OUTW + idx] + c);
  ctx[idx] = c;
  if (idx >= 15 * DMODEL) h15[idx - 15 * DMODEL] = (_Float16)c;
}

// ---------------- generation GEMVs ----------------
__global__ void k_layer_gemv(const _Float16* __restrict__ hbf,
                             const _Float16* __restrict__ WlB,
                             float* __restrict__ ctx, _Float16* __restrict__ h15) {
  int wave = blockIdx.x * (blockDim.x >> 5) + (threadIdx.x >> 5); // 1024 waves
  int lane = threadIdx.x & 31;
  int l = wave >> 6;
  int n0 = (wave & 63) << 4;
  float acc = gemv16(WlB + ((size_t)l << 20) + n0, DMODEL, hbf);
  if (lane < 16) {
    int g = (l << 10) + n0 + lane;
    float v = tanhf(acc + ctx[g]);
    ctx[g] = v;                          // same thread read->write: safe
    if (l == 15) h15[n0 + lane] = (_Float16)v;
  }
}

__global__ void k_logits_gemv(const _Float16* __restrict__ h15,
                              const _Float16* __restrict__ WoB,
                              float* __restrict__ logits) {
  int wave = blockIdx.x * (blockDim.x >> 5) + (threadIdx.x >> 5); // 2000 waves
  int lane = threadIdx.x & 31;
  int n0 = wave << 4;
  float acc = gemv16(WoB + n0, VOCAB, h15);
  if (lane < 16) logits[n0 + lane] = acc;
}

// ---------------- sampling (threefry2x32 / gumbel-max / top-k) ----------------
__device__ __forceinline__ void threefry2x32(uint32_t k0, uint32_t k1,
                                             uint32_t c0, uint32_t c1,
                                             uint32_t& o0, uint32_t& o1) {
  uint32_t ks2 = 0x1BD11BDAu ^ k0 ^ k1;
  uint32_t x0 = c0 + k0, x1 = c1 + k1;
#define TF_R(d) { x0 += x1; x1 = (x1 << (d)) | (x1 >> (32 - (d))); x1 ^= x0; }
  TF_R(13) TF_R(15) TF_R(26) TF_R(6)  x0 += k1;  x1 += ks2 + 1u;
  TF_R(17) TF_R(29) TF_R(16) TF_R(24) x0 += ks2; x1 += k0 + 2u;
  TF_R(13) TF_R(15) TF_R(26) TF_R(6)  x0 += k0;  x1 += k1 + 3u;
  TF_R(17) TF_R(29) TF_R(16) TF_R(24) x0 += k1;  x1 += ks2 + 4u;
  TF_R(13) TF_R(15) TF_R(26) TF_R(6)  x0 += ks2; x1 += k0 + 5u;
#undef TF_R
  o0 = x0; o1 = x1;
}

__device__ __forceinline__ uint32_t fkey(float f) {         // monotone uint map
  uint32_t u = __float_as_uint(f);
  return (u & 0x80000000u) ? ~u : (u | 0x80000000u);
}
__device__ __forceinline__ float keyinv(uint32_t k) {
  uint32_t u = (k & 0x80000000u) ? (k ^ 0x80000000u) : ~k;
  return __uint_as_float(u);
}
__device__ __forceinline__ float gumbel_bits(uint32_t ka, uint32_t kb, int j) {
  uint32_t o0, o1, bits;
  if (j < VOCAB / 2) { threefry2x32(ka, kb, (uint32_t)j, (uint32_t)(VOCAB / 2 + j), o0, o1); bits = o0; }
  else               { threefry2x32(ka, kb, (uint32_t)(j - VOCAB / 2), (uint32_t)j, o0, o1); bits = o1; }
  float u = __uint_as_float(0x3f800000u | (bits >> 9)) - 1.0f;
  u = fmaxf(u, 1.1754944e-38f);
  return -logf(-logf(u));
}

__global__ void k_sample(const float* __restrict__ logits, float* __restrict__ lbuf,
                         int* __restrict__ counts,
                         const float* __restrict__ temp_p, const float* __restrict__ rep_p,
                         const float* __restrict__ E, _Float16* __restrict__ hbf,
                         long long* __restrict__ tok_out, int step) {
  const int tid = threadIdx.x;                 // single 1024-thread block
  __shared__ int   s_cnt;
  __shared__ float s_v[1024];
  __shared__ int   s_i[1024];
  __shared__ int   s_tok;

  const float invT = 1.0f / temp_p[0];
  const float invR = 1.0f / rep_p[0];

  // pass 1: temperature + repetition penalty
  for (int j = tid; j < VOCAB; j += 1024) {
    float l = logits[j] * invT;
    if (counts[j] > 0) l *= invR;
    lbuf[j] = l;
  }
  __syncthreads();

  // top-k threshold: largest key m with count(key >= m) >= TOPK
  uint32_t lo = 0u, hi = 0xFFFFFFFFu;
  for (int it = 0; it < 32; ++it) {
    if (lo >= hi) break;
    uint32_t mid = lo + ((hi - lo) >> 1) + 1u;
    if (tid == 0) s_cnt = 0;
    __syncthreads();
    int local = 0;
    for (int j = tid; j < VOCAB; j += 1024)
      if (fkey(lbuf[j]) >= mid) local++;
    atomicAdd(&s_cnt, local);
    __syncthreads();
    int cnt = s_cnt;
    __syncthreads();
    if (cnt >= TOPK) lo = mid; else hi = mid - 1u;
  }
  const float kth = keyinv(lo);

  // per-step key: keys = split(key(42), 256); key data = (0, 42)
  uint32_t ka, kb;
  {
    uint32_t o0, o1;
    uint32_t j0 = (uint32_t)(2 * step), j1 = (uint32_t)(2 * step + 1);
    if (j0 < 256u) { threefry2x32(0u, 42u, j0, 256u + j0, o0, o1); ka = o0; }
    else           { threefry2x32(0u, 42u, j0 - 256u, j0, o0, o1); ka = o1; }
    if (j1 < 256u) { threefry2x32(0u, 42u, j1, 256u + j1, o0, o1); kb = o0; }
    else           { threefry2x32(0u, 42u, j1 - 256u, j1, o0, o1); kb = o1; }
  }

  // gumbel-max over masked logits
  float bv = -INFINITY; int bi = 0;
  for (int j = tid; j < VOCAB; j += 1024) {
    float lv = lbuf[j];
    float lf = (lv < kth) ? -1e30f : lv;
    float sc = lf + gumbel_bits(ka, kb, j);
    if (sc > bv) { bv = sc; bi = j; }
  }
  s_v[tid] = bv; s_i[tid] = bi;
  __syncthreads();
  for (int off = 512; off > 0; off >>= 1) {
    if (tid < off) {
      float ov = s_v[tid + off]; int oi = s_i[tid + off];
      if (ov > s_v[tid] || (ov == s_v[tid] && oi < s_i[tid])) { s_v[tid] = ov; s_i[tid] = oi; }
    }
    __syncthreads();
  }
  if (tid == 0) {
    int tok = s_i[0];
    counts[tok] += 1;
    tok_out[step] = (long long)tok;
    s_tok = tok;
  }
  __syncthreads();
  // next-token embedding (f16) for the generation GEMVs
  hbf[tid] = (_Float16)E[(size_t)s_tok * DMODEL + tid];
}

// ---------------- host launch ----------------
extern "C" void kernel_launch(void* const* d_in, const int* in_sizes, int n_in,
                              void* d_out, int out_size, void* d_ws, size_t ws_size,
                              hipStream_t stream) {
  const int*   input_ids = (const int*)d_in[0];
  const float* E         = (const float*)d_in[1];
  const float* Wl        = (const float*)d_in[2];
  const float* Wo        = (const float*)d_in[3];
  const float* mems_init = (const float*)d_in[4];
  const int*   tok_cnt0  = (const int*)d_in[5];
  const float* temp      = (const float*)d_in[6];
  const float* rep       = (const float*)d_in[7];
  (void)in_sizes; (void)n_in; (void)out_size; (void)ws_size;

  // carve workspace (256B aligned)
  char* w = (char*)d_ws;
  auto carve = [&](size_t bytes) -> void* {
    void* p = (void*)w; w += (bytes + 255) & ~(size_t)255; return p;
  };
  _Float16* WlB  = (_Float16*)carve((size_t)NLAYER * DMODEL * DMODEL * 2); // 32 MB
  _Float16* WoB  = (_Float16*)carve((size_t)DMODEL * VOCAB * 2);           // 62 MB
  float*    Z    = (float*)carve((size_t)TPROMPT * OUTW * 4);              // 8 MB
  float*    ctx  = (float*)carve((size_t)OUTW * 4);
  float*    lg   = (float*)carve((size_t)VOCAB * 4);
  float*    lbuf = (float*)carve((size_t)VOCAB * 4);
  int*      cnts = (int*)carve((size_t)VOCAB * 4);
  _Float16* H    = (_Float16*)carve((size_t)TPROMPT * DMODEL * 2);
  _Float16* h15  = (_Float16*)carve((size_t)DMODEL * 2);
  _Float16* hbf  = (_Float16*)carve((size_t)DMODEL * 2);
  long long* toks = (long long*)d_out;                                     // (1,256) int64

  // weights -> f16 (both fit in 192MB L2 afterwards: 96 MB total)
  k_f32_to_f16<<<8192, 256, 0, stream>>>(Wl, WlB, NLAYER * DMODEL * DMODEL);
  k_f32_to_f16<<<8192, 256, 0, stream>>>(Wo, WoB, DMODEL * VOCAB);
  k_init_state<<<(VOCAB + 255) / 256, 256, 0, stream>>>(mems_init, tok_cnt0, ctx, cnts);

  // ---- prompt phase: one big WMMA GEMM + elementwise scan ----
  k_build_H<<<(TPROMPT * DMODEL) / 256, 256, 0, stream>>>(input_ids, E, H);
  k_prompt_gemm<<<1024, 256, 0, stream>>>(H, WlB, Z);      // 8192 waves
  k_prompt_scan<<<OUTW / 1024, 1024, 0, stream>>>(Z, ctx, h15);
  k_logits_gemv<<<250, 256, 0, stream>>>(h15, WoB, lg);    // pred_logits

  // ---- generation: sample -> 16 layer GEMVs -> logits GEMV ----
  for (int i = 0; i < NGEN; ++i) {
    k_sample<<<1, 1024, 0, stream>>>(lg, lbuf, cnts, temp, rep, E, hbf, toks, i);
    if (i < NGEN - 1) {
      k_layer_gemv<<<128, 256, 0, stream>>>(hbf, WlB, ctx, h15);   // 1024 waves
      k_logits_gemv<<<250, 256, 0, stream>>>(h15, WoB, lg);        // 2000 waves
    }
  }
}
